// BigBirdAttention_65687229825639
// MI455X (gfx1250) — matrix operand order repaired and checked by reading, not compiled, not run on hardware
//
#include <hip/hip_runtime.h>

// ---------------------------------------------------------------------------
// BigBird attention for MI455X (gfx1250), wave32, WMMA bf16.
//
// Pipeline (all on `stream`):
//   1) convert x -> bf16; pack W{q,k,v,o} transposed+pair-interleaved bf16
//   2) build per-16x16-tile mask bitmaps (C-fragment lane order) + any-flags
//   3) GEMM (WMMA bf16, 2x2 register-blocked) -> Q,K (B,H,T,HD) bf16,
//      V transposed (B,H,HD,T) bf16
//   4) flash attention, one wave per (b,h,16-query-rows), masked-tile skipping
//   5) GEMM (WMMA bf16) ctx @ WoT + bo -> f32 output (non-temporal stores)
// ---------------------------------------------------------------------------

typedef __attribute__((ext_vector_type(16))) __bf16 v16bf;
typedef __attribute__((ext_vector_type(8)))  float  v8f;

union Frag {
  v16bf v;
  uint4 u4[2];
  unsigned int u32[8];
};

static __device__ __forceinline__ unsigned short f2bf(float f) {
  unsigned int u = __float_as_uint(f);
  u += 0x7FFFu + ((u >> 16) & 1u);           // round-to-nearest-even
  return (unsigned short)(u >> 16);
}

constexpr int Bdim = 2, T = 2048, D = 768, H = 12, HD = 64;
constexpr int BT = Bdim * T;    // 4096 rows for the projections
constexpr int QT = T / 16;      // 128 token tiles
constexpr int MT2 = BT / 32;    // 128 super-tiles (32 rows each)
constexpr int NT2 = D / 32;     // 24 super-tiles (32 cols each)
constexpr float NEG = -1e30f;

// --- 1a) x (f32) -> bf16 ---------------------------------------------------
__global__ __launch_bounds__(256) void convert_x(const float* __restrict__ x,
                                                 unsigned short* __restrict__ xb) {
  int i = blockIdx.x * 256 + threadIdx.x;    // grid sized exactly BT*D/256
  xb[i] = f2bf(x[i]);
}

// --- 1b) W (D,D) f32 -> pre-transposed, pair-interleaved bf16 --------------
// B2[(k>>1)*(2D) + 2n + (k&1)] = W[n*D + k]  (= W.T element (k,n))
__global__ __launch_bounds__(256) void pack_w(const float* __restrict__ W,
                                              unsigned short* __restrict__ B2) {
  int i = blockIdx.x * 256 + threadIdx.x;    // grid sized exactly D*D/256
  int k = i % D, n = i / D;                  // k fastest -> coalesced reads
  B2[(size_t)(k >> 1) * (2 * D) + 2 * n + (k & 1)] = f2bf(W[(size_t)n * D + k]);
}

// --- 2) mask tile bitmaps in C-fragment lane order -------------------------
// C layout: lane l owns (row = e + 8*(l>>4), col = l&15), e = 0..7.
__global__ __launch_bounds__(256) void build_maskbits(const unsigned char* __restrict__ mask,
                                                      unsigned char* __restrict__ mbits,
                                                      unsigned char* __restrict__ flags) {
  int wv = threadIdx.x >> 5, lane = threadIdx.x & 31;
  int tile = blockIdx.x * 8 + wv;            // QT*QT tiles, 8 per block
  int tr = tile >> 7, tc = tile & (QT - 1);
  int ln = lane & 15, half = lane >> 4;
  unsigned byte = 0;
#pragma unroll
  for (int e = 0; e < 8; ++e) {
    int r = tr * 16 + e + 8 * half, c = tc * 16 + ln;
    if (mask[(size_t)r * T + c]) byte |= 1u << e;
  }
  mbits[(size_t)tile * 32 + lane] = (unsigned char)byte;
  unsigned long long any = __ballot(byte != 0);
  if (lane == 0) flags[tile] = any ? 1 : 0;
}

// --- 3/5) GEMM: out = A(4096xD bf16) @ W.T + bias --------------------------
// 2x2 register blocking: one wave owns a 32x32 output block (4 WMMA tiles),
// reusing each A/B fragment twice -> 20 loads per 4 WMMAs.
// mode 0: f32 row-major out (non-temporal)   mode 1: Q bf16 (B,H,T,HD)
// mode 2: K bf16 (B,H,T,HD)                  mode 3: V bf16 transposed (B,H,HD,T)
__global__ __launch_bounds__(256) void gemm_bias_wmma(const unsigned short* __restrict__ A,
                                                      const unsigned short* __restrict__ B2,
                                                      const float* __restrict__ bias,
                                                      void* __restrict__ outp, int mode) {
  const int wv = threadIdx.x >> 5, lane = threadIdx.x & 31;
  const int tile = blockIdx.x * 8 + wv;      // MT2*NT2 super-tiles, 8 per block
  const int mt = tile / NT2, nt = tile % NT2;
  const int ln = lane & 15, half = lane >> 4;

  const unsigned short* Arow0 = A + (size_t)(mt * 32 + ln) * D;        // rows ln, ln+16
  const unsigned short* Arow1 = Arow0 + (size_t)16 * D;
  const size_t bcol0 = 2 * (size_t)(nt * 32 + ln);                     // cols ln, ln+16
  v8f acc00 = {}, acc01 = {}, acc10 = {}, acc11 = {};

  for (int k0 = 0; k0 < D; k0 += 32) {
    // latency hiding for the first (HBM) pass; later passes hit L2 (192 MB)
    __builtin_prefetch(Arow0 + k0 + 64, 0, 1);
    __builtin_prefetch(Arow1 + k0 + 64, 0, 1);

    Frag a0, a1, b0, b1;
    a0.u4[0] = *(const uint4*)(Arow0 + k0 + half * 8);
    a0.u4[1] = *(const uint4*)(Arow0 + k0 + 16 + half * 8);
    a1.u4[0] = *(const uint4*)(Arow1 + k0 + half * 8);
    a1.u4[1] = *(const uint4*)(Arow1 + k0 + 16 + half * 8);
#pragma unroll
    for (int i = 0; i < 8; ++i) {
      const unsigned short* bp = B2 + (size_t)((k0 >> 1) + (i >> 2) * 8 + half * 4 + (i & 3)) * (2 * D);
      b0.u32[i] = *(const unsigned int*)(bp + bcol0);
      b1.u32[i] = *(const unsigned int*)(bp + bcol0 + 32);
    }
    acc00 = __builtin_amdgcn_wmma_f32_16x16x32_bf16(false, a0.v, false, b0.v, (short)0, acc00, false, false);
    acc01 = __builtin_amdgcn_wmma_f32_16x16x32_bf16(false, a0.v, false, b1.v, (short)0, acc01, false, false);
    acc10 = __builtin_amdgcn_wmma_f32_16x16x32_bf16(false, a1.v, false, b0.v, (short)0, acc10, false, false);
    acc11 = __builtin_amdgcn_wmma_f32_16x16x32_bf16(false, a1.v, false, b1.v, (short)0, acc11, false, false);
  }

#pragma unroll
  for (int mi = 0; mi < 2; ++mi) {
#pragma unroll
    for (int ni = 0; ni < 2; ++ni) {
      const v8f& acc = mi ? (ni ? acc11 : acc10) : (ni ? acc01 : acc00);
      const int col = nt * 32 + ni * 16 + ln;
      const float bv = bias[col];
#pragma unroll
      for (int e = 0; e < 8; ++e) {
        int row = mt * 32 + mi * 16 + e + 8 * half;   // token row 0..4095
        float val = acc[e] + bv;
        if (mode == 0) {
          __builtin_nontemporal_store(val, (float*)outp + (size_t)row * D + col);
        } else {
          int bi = row >> 11, t = row & (T - 1);
          int h = col >> 6, hd = col & 63;
          unsigned short us = f2bf(val);
          if (mode == 3)
            ((unsigned short*)outp)[((size_t)(bi * H + h) * HD + hd) * T + t] = us;
          else
            ((unsigned short*)outp)[((size_t)(bi * H + h) * T + t) * HD + hd] = us;
        }
      }
    }
  }
}

// --- 4) flash attention: one wave per (b,h,qtile) --------------------------
__global__ __launch_bounds__(256) void bigbird_attn_wmma(
    const unsigned short* __restrict__ Q, const unsigned short* __restrict__ Kh,
    const unsigned short* __restrict__ Vt, const unsigned char* __restrict__ mbits,
    const unsigned char* __restrict__ flags, unsigned short* __restrict__ ctx) {
  __shared__ __align__(16) unsigned short ldsP[8][16][32];  // per-wave P tile

  const int wv = threadIdx.x >> 5, lane = threadIdx.x & 31;
  const int w = blockIdx.x * 8 + wv;         // B*H*QT waves
  const int qt = w & (QT - 1);
  const int bh = w >> 7;                     // 0..23
  const int ln = lane & 15, half = lane >> 4;

  const unsigned short* Qb = Q  + ((size_t)bh * T + qt * 16) * HD;
  const unsigned short* Kb = Kh + (size_t)bh * T * HD;
  const unsigned short* Vb = Vt + (size_t)bh * HD * T;

  // Q A-frags (hd 0..31, 32..63): 2 x b128 per frag per lane
  Frag qa[2];
#pragma unroll
  for (int f = 0; f < 2; ++f) {
    const unsigned short* p = Qb + ln * HD + f * 32 + half * 16;
    qa[f].u4[0] = *(const uint4*)(p);
    qa[f].u4[1] = *(const uint4*)(p + 8);
  }

  float mrow[8], lrow[8];
  v8f o[4] = {};
#pragma unroll
  for (int e = 0; e < 8; ++e) { mrow[e] = NEG; lrow[e] = 0.f; }

  for (int kt = 0; kt < QT; kt += 2) {       // 32 keys per chunk
    if ((flags[qt * QT + kt] | flags[qt * QT + kt + 1]) == 0) continue;

    // prefetch next chunk's K rows while we compute this one
    if (kt + 2 < QT) {
      __builtin_prefetch(Kb + (size_t)((kt + 2) * 16 + ln) * HD, 0, 1);
      __builtin_prefetch(Vb + (size_t)ln * T + (kt + 2) * 16, 0, 1);
    }

    float s[2][8];
#pragma unroll
    for (int sub = 0; sub < 2; ++sub) {
      const int key0 = (kt + sub) * 16;
      // K B-frags: lane holds key column key0+ln; hd contiguous -> 2 x b128
      Frag kb0, kb1;
      const unsigned short* kp = Kb + (size_t)(key0 + ln) * HD + half * 16;
      kb0.u4[0] = *(const uint4*)(kp);      kb0.u4[1] = *(const uint4*)(kp + 8);
      kb1.u4[0] = *(const uint4*)(kp + 32); kb1.u4[1] = *(const uint4*)(kp + 40);
      v8f c = {};
      c = __builtin_amdgcn_wmma_f32_16x16x32_bf16(false, qa[0].v, false, kb0.v, (short)0, c, false, false);
      c = __builtin_amdgcn_wmma_f32_16x16x32_bf16(false, qa[1].v, false, kb1.v, (short)0, c, false, false);
      unsigned mb = mbits[(size_t)(qt * QT + kt + sub) * 32 + lane];
#pragma unroll
      for (int e = 0; e < 8; ++e)
        s[sub][e] = (mb >> e & 1u) ? c[e] * 0.125f : NEG;   // scale = HD^-0.5
    }

    // online softmax update; rows grouped in 16-lane halves
#pragma unroll
    for (int e = 0; e < 8; ++e) {
      float tm = fmaxf(s[0][e], s[1][e]);
      tm = fmaxf(tm, __shfl_xor(tm, 1, 16));
      tm = fmaxf(tm, __shfl_xor(tm, 2, 16));
      tm = fmaxf(tm, __shfl_xor(tm, 4, 16));
      tm = fmaxf(tm, __shfl_xor(tm, 8, 16));
      float mnew = fmaxf(mrow[e], tm);
      float alpha = __expf(mrow[e] - mnew);
      float p0 = (s[0][e] > -1e29f) ? __expf(s[0][e] - mnew) : 0.f;
      float p1 = (s[1][e] > -1e29f) ? __expf(s[1][e] - mnew) : 0.f;
      float ps = p0 + p1;
      ps += __shfl_xor(ps, 1, 16);
      ps += __shfl_xor(ps, 2, 16);
      ps += __shfl_xor(ps, 4, 16);
      ps += __shfl_xor(ps, 8, 16);
      lrow[e] = lrow[e] * alpha + ps;
      mrow[e] = mnew;
#pragma unroll
      for (int g = 0; g < 4; ++g) o[g][e] *= alpha;
      int row = e + 8 * half;                // C-layout row owned by this lane
      ldsP[wv][row][ln]      = f2bf(p0);
      ldsP[wv][row][16 + ln] = f2bf(p1);
    }

    // fence: LDS stores (cross-lane) must land before A-frag reads
    asm volatile("s_wait_dscnt 0" ::: "memory");

    // P A-frag from LDS (16x32): 2 x ds_load_b128 per lane
    Frag pa;
    pa.u4[0] = *(const uint4*)&ldsP[wv][ln][half * 16];
    pa.u4[1] = *(const uint4*)&ldsP[wv][ln][half * 16 + 8];

    const int keyb = kt * 16;
#pragma unroll
    for (int g = 0; g < 4; ++g) {
      // V B-frags from transposed V: keys contiguous -> 2 x b128
      Frag vb;
      const unsigned short* vp = Vb + (size_t)(g * 16 + ln) * T + keyb + half * 16;
      vb.u4[0] = *(const uint4*)(vp);
      vb.u4[1] = *(const uint4*)(vp + 8);
      o[g] = __builtin_amdgcn_wmma_f32_16x16x32_bf16(false, pa.v, false, vb.v,
                                                     (short)0, o[g], false, false);
    }
    asm volatile("" ::: "memory");           // keep next-iter stores below loads
  }

  // epilogue: normalize, write ctx bf16 (row-major 4096 x 768)
  const int b = bh / H, h = bh % H;
#pragma unroll
  for (int e = 0; e < 8; ++e) {
    float inv = (lrow[e] > 1e-20f) ? (1.f / lrow[e]) : 0.f;
    int bt = b * T + qt * 16 + e + 8 * half;
#pragma unroll
    for (int g = 0; g < 4; ++g)
      ctx[(size_t)bt * D + h * HD + g * 16 + ln] = f2bf(o[g][e] * inv);
  }
}

// ---------------------------------------------------------------------------
extern "C" void kernel_launch(void* const* d_in, const int* in_sizes, int n_in,
                              void* d_out, int out_size, void* d_ws, size_t ws_size,
                              hipStream_t stream) {
  const float* x  = (const float*)d_in[0];
  const float* Wq = (const float*)d_in[1];
  const float* bq = (const float*)d_in[2];
  const float* Wk = (const float*)d_in[3];
  const float* bk = (const float*)d_in[4];
  const float* Wv = (const float*)d_in[5];
  const float* bv = (const float*)d_in[6];
  const float* Wo = (const float*)d_in[7];
  const float* bo = (const float*)d_in[8];
  const unsigned char* mask = (const unsigned char*)d_in[9];
  float* out = (float*)d_out;

  char* ws = (char*)d_ws;
  size_t off = 0;
  auto take = [&](size_t bytes) { char* p = ws + off; off += (bytes + 255) & ~size_t(255); return p; };
  unsigned short* xbf  = (unsigned short*)take((size_t)BT * D * 2);   // 6.0 MB
  unsigned short* wq2  = (unsigned short*)take((size_t)D * D * 2);
  unsigned short* wk2  = (unsigned short*)take((size_t)D * D * 2);
  unsigned short* wv2  = (unsigned short*)take((size_t)D * D * 2);
  unsigned short* wo2  = (unsigned short*)take((size_t)D * D * 2);
  unsigned short* Qh   = (unsigned short*)take((size_t)BT * D * 2);
  unsigned short* Kh   = (unsigned short*)take((size_t)BT * D * 2);
  unsigned short* Vth  = (unsigned short*)take((size_t)BT * D * 2);
  unsigned short* ctx  = (unsigned short*)take((size_t)BT * D * 2);
  unsigned char*  mbit = (unsigned char*) take((size_t)QT * QT * 32);
  unsigned char*  flg  = (unsigned char*) take((size_t)QT * QT);      // total ~37 MB

  convert_x<<<BT * D / 256, 256, 0, stream>>>(x, xbf);
  pack_w<<<D * D / 256, 256, 0, stream>>>(Wq, wq2);
  pack_w<<<D * D / 256, 256, 0, stream>>>(Wk, wk2);
  pack_w<<<D * D / 256, 256, 0, stream>>>(Wv, wv2);
  pack_w<<<D * D / 256, 256, 0, stream>>>(Wo, wo2);
  build_maskbits<<<QT * QT / 8, 256, 0, stream>>>(mask, mbit, flg);

  gemm_bias_wmma<<<MT2 * NT2 / 8, 256, 0, stream>>>(xbf, wq2, bq, Qh, 1);
  gemm_bias_wmma<<<MT2 * NT2 / 8, 256, 0, stream>>>(xbf, wk2, bk, Kh, 2);
  gemm_bias_wmma<<<MT2 * NT2 / 8, 256, 0, stream>>>(xbf, wv2, bv, Vth, 3);

  bigbird_attn_wmma<<<Bdim * H * QT / 8, 256, 0, stream>>>(Qh, Kh, Vth, mbit, flg, ctx);

  gemm_bias_wmma<<<MT2 * NT2 / 8, 256, 0, stream>>>(ctx, wo2, bo, out, 0);
}
